// TransformerBlock_730144440854
// MI455X (gfx1250) — matrix operand order repaired and checked by reading, not compiled, not run on hardware
//
#include <hip/hip_runtime.h>
#include <hip/hip_bf16.h>

// ---------------------------------------------------------------------------
// Types for CDNA5 WMMA (wave32, V_WMMA_F32_16X16X32_BF16)
// ---------------------------------------------------------------------------
typedef __attribute__((ext_vector_type(16))) __bf16 v16bf;
typedef __attribute__((ext_vector_type(8)))  float  v8f;

union BF16Frag { unsigned int u[8]; v16bf v; };

__device__ __forceinline__ unsigned short f2bf(float f) {
  union { float f; unsigned int u; } v; v.f = f;
  unsigned int r = v.u + 0x7FFFu + ((v.u >> 16) & 1u);   // round-nearest-even
  return (unsigned short)(r >> 16);
}

__device__ __forceinline__ v8f wmma_bf16(BF16Frag a, BF16Frag b, v8f c) {
  // D = A(16x32 bf16) * B(32x16 bf16) + C(16x16 f32)
  return __builtin_amdgcn_wmma_f32_16x16x32_bf16(
      false, a.v, false, b.v, (short)0, c, false, false);
}

// DPP16 ROW_XMASK butterfly shuffle within each 16-lane row: pure VALU,
// no LDS round-trip (replaces ds_bpermute-based __shfl_xor).
template <int XMASK>
__device__ __forceinline__ float dpp_xmask(float v) {
  int i = __float_as_int(v);
  int p = __builtin_amdgcn_update_dpp(i, i, 0x160 + XMASK, 0xF, 0xF, true);
  return __int_as_float(p);
}

__device__ __forceinline__ float row16_max(float v) {
  v = fmaxf(v, dpp_xmask<1>(v));
  v = fmaxf(v, dpp_xmask<2>(v));
  v = fmaxf(v, dpp_xmask<4>(v));
  v = fmaxf(v, dpp_xmask<8>(v));
  return v;
}

__device__ __forceinline__ float row16_sum(float v) {
  v += dpp_xmask<1>(v);
  v += dpp_xmask<2>(v);
  v += dpp_xmask<4>(v);
  v += dpp_xmask<8>(v);
  return v;
}

// Problem constants: B=8, C=256, D=3, N=1024, NC=3N=3072, H=8, Ch=32, F=96.

// ---------------------------------------------------------------------------
// f32 -> bf16 converter
// ---------------------------------------------------------------------------
__global__ __launch_bounds__(256) void k_f32_to_bf16(
    const float* __restrict__ in, unsigned short* __restrict__ out, int n) {
  int i = blockIdx.x * 256 + threadIdx.x;
  if (i < n) out[i] = f2bf(in[i]);
}

// ---------------------------------------------------------------------------
// u[o,d] = eps * b[o,d] / ||b[o,:]||   (256 channels, 3 dims)
// ---------------------------------------------------------------------------
__global__ __launch_bounds__(256) void k_bias_dir(
    const float* __restrict__ b3, float* __restrict__ u) {
  int o = threadIdx.x;
  float b0 = b3[o * 3 + 0], b1 = b3[o * 3 + 1], b2 = b3[o * 3 + 2];
  float inv = 1e-6f / sqrtf(b0 * b0 + b1 * b1 + b2 * b2);
  u[o * 3 + 0] = b0 * inv;
  u[o * 3 + 1] = b1 * inv;
  u[o * 3 + 2] = b2 * inv;
}

// ---------------------------------------------------------------------------
// Tiled bf16 WMMA GEMM:  Out[b][m][col] = sum_k W[k*M+m] * X[b][k][col] (+u[m, col/1024])
// W: f32 [K x M] (converted to bf16 on stage-in), X: bf16 [batch][K][NC]
// Block: 256 threads = 8 waves (4 m-waves x 2 n-waves). Tile 128x128, K-step 32.
// Templated on output dtype / bias presence so the epilogue is branch-free.
// ---------------------------------------------------------------------------
template <int OUT_BF16, int HAS_BIAS>
__global__ __launch_bounds__(256) void k_gemm_vn(
    const float* __restrict__ W, const unsigned short* __restrict__ X,
    void* __restrict__ Out, const float* __restrict__ u,
    int M, int K, int NC) {
  __shared__ __align__(8) unsigned short sA[128 * 34];  // [m][k], pad 34
  __shared__ __align__(8) unsigned short sX[128 * 34];  // [n][k], pad 34
  const int b = blockIdx.z;
  const int m0 = blockIdx.y * 128, n0 = blockIdx.x * 128;
  const int tid = threadIdx.x;
  const int lane = tid & 31, wave = tid >> 5;
  const int wm = wave & 3, wn = wave >> 2;
  const int hl = lane >> 4, l15 = lane & 15;
  const unsigned short* Xb = X + (size_t)b * K * NC;

  v8f acc[2][4];
#pragma unroll
  for (int i = 0; i < 2; ++i)
#pragma unroll
    for (int j = 0; j < 4; ++j)
#pragma unroll
      for (int r = 0; r < 8; ++r) acc[i][j][r] = 0.f;

  const int kl = tid >> 3;
  const int ms = (tid & 7) * 16;

  for (int k0 = 0; k0 < K; k0 += 32) {
    // Cooperative stage: each thread owns one k-lane, 16 contiguous m / n.
    {
      const float* srcW = W + (size_t)(k0 + kl) * M + (m0 + ms);
      const unsigned short* srcX = Xb + (size_t)(k0 + kl) * NC + (n0 + ms);
#pragma unroll
      for (int i = 0; i < 16; ++i) {
        sA[(ms + i) * 34 + kl] = f2bf(srcW[i]);
        sX[(ms + i) * 34 + kl] = srcX[i];
      }
      // Speculative prefetch of the next K-slab (global_prefetch_b8).
      if (k0 + 32 < K) {
        __builtin_prefetch(W + (size_t)(k0 + 32 + kl) * M + (m0 + ms), 0, 0);
        __builtin_prefetch(Xb + (size_t)(k0 + 32 + kl) * NC + (n0 + ms), 0, 0);
      }
    }
    __syncthreads();

    BF16Frag af[2], bfr[4];
#pragma unroll
    for (int mt = 0; mt < 2; ++mt) {
      int m = wm * 32 + mt * 16 + l15;
#pragma unroll
      for (int v = 0; v < 8; ++v) {
        int kk = ((v >> 2) * 16) + hl * 8 + ((v & 3) * 2);  // A-layout
        af[mt].u[v] = *(const unsigned int*)&sA[m * 34 + kk];
      }
    }
#pragma unroll
    for (int nt = 0; nt < 4; ++nt) {
      int n = wn * 64 + nt * 16 + l15;
#pragma unroll
      for (int v = 0; v < 8; ++v) {
        int kk = ((v >> 2) * 16) + hl * 8 + ((v & 3) * 2);  // B-layout
        bfr[nt].u[v] = *(const unsigned int*)&sX[n * 34 + kk];
      }
    }
#pragma unroll
    for (int mt = 0; mt < 2; ++mt)
#pragma unroll
      for (int nt = 0; nt < 4; ++nt)
        acc[mt][nt] = wmma_bf16(af[mt], bfr[nt], acc[mt][nt]);
    __syncthreads();
  }

  // Epilogue: C/D layout -> global, optional VN bias u[m, d] with d = col/1024.
#pragma unroll
  for (int mt = 0; mt < 2; ++mt) {
#pragma unroll
    for (int nt = 0; nt < 4; ++nt) {
      int col = n0 + wn * 64 + nt * 16 + l15;
      int dd = col >> 10;
#pragma unroll
      for (int r = 0; r < 8; ++r) {
        int row = m0 + wm * 32 + mt * 16 + hl * 8 + r;
        float val = acc[mt][nt][r];
        if (HAS_BIAS) val += u[row * 3 + dd];
        size_t off = (size_t)b * M * NC + (size_t)row * NC + col;
        if (OUT_BF16) ((unsigned short*)Out)[off] = f2bf(val);
        else          ((float*)Out)[off] = val;
      }
    }
  }
}

// ---------------------------------------------------------------------------
// Fused flash-style VN attention per (b, head, 64-row M block).
// q,k,z: bf16 [B][C=256][3N=3072]; per head: F = Ch*3 = 96 feature rows.
// 4 waves; each wave owns one 16-row M tile and loops N in 32-col chunks:
//   S = Q*K^T (3 WMMA K-steps x 2 col tiles), online softmax (DPP row_xmask
//   butterflies, no LDS), O += P*V (6 tiles).
// ---------------------------------------------------------------------------
__global__ __launch_bounds__(128) void k_flash_attn(
    const unsigned short* __restrict__ qb,
    const unsigned short* __restrict__ kb,
    const unsigned short* __restrict__ zb,
    unsigned short* __restrict__ ob) {
  __shared__ __align__(8) unsigned short Klds[32 * 98];      // [n][f]
  __shared__ __align__(8) unsigned short Zlds[96 * 34];      // [f][n]
  __shared__ __align__(8) unsigned short Plds[4 * 16 * 34];  // per-wave P
  const int b = blockIdx.z, h = blockIdx.y;
  const int m0 = blockIdx.x * 64;
  const int tid = threadIdx.x;
  const int lane = tid & 31, wave = tid >> 5;
  const int hl = lane >> 4, l15 = lane & 15;
  const float scale = 0.102062072616f;  // 1/sqrt(3*Ch) = 1/sqrt(96)
  const size_t base = ((size_t)(b * 256 + h * 32)) * 3072;

  // Load resident Q A-fragments (3 K-steps of 32 over F=96).
  BF16Frag qf[3];
  {
    const int m = m0 + wave * 16 + l15;
#pragma unroll
    for (int s = 0; s < 3; ++s) {
#pragma unroll
      for (int v = 0; v < 8; ++v) {
        int f0 = 32 * s + ((v >> 2) * 16) + hl * 8 + ((v & 3) * 2);
        int f1 = f0 + 1;
        unsigned int lo = qb[base + (size_t)(f0 / 3) * 3072 + (f0 % 3) * 1024 + m];
        unsigned int hi = qb[base + (size_t)(f1 / 3) * 3072 + (f1 % 3) * 1024 + m];
        qf[s].u[v] = lo | (hi << 16);
      }
    }
  }

  float mstate[8], lstate[8];
  v8f o[6];
#pragma unroll
  for (int r = 0; r < 8; ++r) { mstate[r] = -3.0e38f; lstate[r] = 0.f; }
#pragma unroll
  for (int t = 0; t < 6; ++t)
#pragma unroll
    for (int r = 0; r < 8; ++r) o[t][r] = 0.f;

  unsigned short* Pw = Plds + wave * 16 * 34;

  for (int nc = 0; nc < 1024; nc += 32) {
    // Stage K chunk as [n][f] and Z chunk as [f][n] (WMMA-pair friendly).
    for (int i = tid; i < 3072; i += 128) {
      int f = i >> 5, n = i & 31;
      size_t g = base + (size_t)(f / 3) * 3072 + (f % 3) * 1024 + (nc + n);
      Klds[n * 98 + f] = kb[g];
      Zlds[f * 34 + n] = zb[g];
    }
    __syncthreads();

    // Scores: two 16x16 tiles (cols nc..nc+15, nc+16..nc+31).
    v8f s0, s1;
#pragma unroll
    for (int r = 0; r < 8; ++r) { s0[r] = 0.f; s1[r] = 0.f; }
#pragma unroll
    for (int s = 0; s < 3; ++s) {
      BF16Frag k0f, k1f;
#pragma unroll
      for (int v = 0; v < 8; ++v) {
        int f = 32 * s + ((v >> 2) * 16) + hl * 8 + ((v & 3) * 2);
        k0f.u[v] = *(const unsigned int*)&Klds[l15 * 98 + f];
        k1f.u[v] = *(const unsigned int*)&Klds[(16 + l15) * 98 + f];
      }
      s0 = wmma_bf16(qf[s], k0f, s0);
      s1 = wmma_bf16(qf[s], k1f, s1);
    }

    // Online softmax: row stats live per (lane-half, accum index r).
    // 16-lane row reductions are DPP ROW_XMASK butterflies (pure VALU).
    float resc[8];
#pragma unroll
    for (int r = 0; r < 8; ++r) {
      float a0 = s0[r] * scale, a1 = s1[r] * scale;
      float mx = row16_max(fmaxf(a0, a1));
      float nm = fmaxf(mstate[r], mx);
      float rs = __expf(mstate[r] - nm);
      float p0 = __expf(a0 - nm);
      float p1 = __expf(a1 - nm);
      float sm = row16_sum(p0 + p1);
      lstate[r] = lstate[r] * rs + sm;
      mstate[r] = nm;
      resc[r] = rs;
      s0[r] = p0;
      s1[r] = p1;
    }
#pragma unroll
    for (int t = 0; t < 6; ++t)
#pragma unroll
      for (int r = 0; r < 8; ++r) o[t][r] *= resc[r];

    // P (C/D layout) -> wave-private LDS -> A-layout fragment.
#pragma unroll
    for (int r = 0; r < 8; ++r) {
      int row = hl * 8 + r;
      Pw[row * 34 + l15]      = f2bf(s0[r]);
      Pw[row * 34 + 16 + l15] = f2bf(s1[r]);
    }
    __syncthreads();

    BF16Frag pf;
#pragma unroll
    for (int v = 0; v < 8; ++v) {
      int k = ((v >> 2) * 16) + hl * 8 + ((v & 3) * 2);
      pf.u[v] = *(const unsigned int*)&Pw[l15 * 34 + k];
    }
    // O += P * V  (V = z^T: 6 column tiles of 16 over F=96)
#pragma unroll
    for (int t = 0; t < 6; ++t) {
      BF16Frag vf;
#pragma unroll
      for (int v = 0; v < 8; ++v) {
        int nl = ((v >> 2) * 16) + hl * 8 + ((v & 3) * 2);
        vf.u[v] = *(const unsigned int*)&Zlds[(t * 16 + l15) * 34 + nl];
      }
      o[t] = wmma_bf16(pf, vf, o[t]);
    }
    __syncthreads();
  }

  // Normalize and scatter out (bf16, [B][C][3N] layout).
#pragma unroll
  for (int t = 0; t < 6; ++t) {
#pragma unroll
    for (int r = 0; r < 8; ++r) {
      float val = o[t][r] / lstate[r];
      int fp = t * 16 + l15;
      int row = m0 + wave * 16 + hl * 8 + r;
      ob[base + (size_t)(fp / 3) * 3072 + (fp % 3) * 1024 + row] = f2bf(val);
    }
  }
}

// ---------------------------------------------------------------------------
// VN LayerNorm over channels of the vector norms, + residual.
// One block per (b, n); thread c handles one channel.
// ---------------------------------------------------------------------------
template <int HAS_OUTB>
__global__ __launch_bounds__(256) void k_vn_ln_res(
    const float* __restrict__ x, const float* __restrict__ res,
    const float* __restrict__ g, const float* __restrict__ be,
    float* __restrict__ out, unsigned short* __restrict__ outb) {
  int n = blockIdx.x, b = blockIdx.y, c = threadIdx.x;
  size_t base = ((size_t)(b * 256 + c)) * 3072 + n;
  float x0 = x[base], x1 = x[base + 1024], x2 = x[base + 2048];
  float nr = sqrtf(x0 * x0 + x1 * x1 + x2 * x2) + 1e-6f;
  __shared__ float ss[256], sq[256];
  ss[c] = nr; sq[c] = nr * nr;
  __syncthreads();
  for (int st = 128; st > 0; st >>= 1) {
    if (c < st) { ss[c] += ss[c + st]; sq[c] += sq[c + st]; }
    __syncthreads();
  }
  float mu = ss[0] * (1.f / 256.f);
  float var = sq[0] * (1.f / 256.f) - mu * mu;
  float ln = (nr - mu) * rsqrtf(var + 1e-5f) * g[c] + be[c];
  float sc = ln / nr;
  float y0 = x0 * sc + res[base];
  float y1 = x1 * sc + res[base + 1024];
  float y2 = x2 * sc + res[base + 2048];
  out[base] = y0; out[base + 1024] = y1; out[base + 2048] = y2;
  if (HAS_OUTB) {
    outb[base] = f2bf(y0); outb[base + 1024] = f2bf(y1); outb[base + 2048] = f2bf(y2);
  }
}

// ---------------------------------------------------------------------------
// VN BatchNorm (training mode): per-channel stats over (B, N) of vector norms.
// ---------------------------------------------------------------------------
__global__ __launch_bounds__(256) void k_vn_bn_stats(
    const float* __restrict__ x, float* __restrict__ stats) {
  int c = blockIdx.x, tid = threadIdx.x;
  float s = 0.f, q = 0.f;
  for (int i = tid; i < 8192; i += 256) {
    int b = i >> 10, n = i & 1023;
    size_t base = ((size_t)(b * 256 + c)) * 3072 + n;
    float x0 = x[base], x1 = x[base + 1024], x2 = x[base + 2048];
    float nr = sqrtf(x0 * x0 + x1 * x1 + x2 * x2) + 1e-6f;
    s += nr; q += nr * nr;
  }
  __shared__ float ss[256], sq[256];
  ss[tid] = s; sq[tid] = q;
  __syncthreads();
  for (int st = 128; st > 0; st >>= 1) {
    if (tid < st) { ss[tid] += ss[tid + st]; sq[tid] += sq[tid + st]; }
    __syncthreads();
  }
  if (tid == 0) {
    float mu = ss[0] * (1.f / 8192.f);
    float var = sq[0] * (1.f / 8192.f) - mu * mu;
    stats[c] = mu;
    stats[256 + c] = rsqrtf(var + 1e-5f);
  }
}

__global__ __launch_bounds__(256) void k_vn_bn_apply(
    const float* __restrict__ x, const float* __restrict__ stats,
    const float* __restrict__ g, const float* __restrict__ be,
    float* __restrict__ out, unsigned short* __restrict__ outb) {
  int idx = blockIdx.x * 256 + threadIdx.x;  // over B*C*N
  int n = idx & 1023;
  int bc = idx >> 10;
  int c = bc & 255;
  size_t base = (size_t)bc * 3072 + n;
  float x0 = x[base], x1 = x[base + 1024], x2 = x[base + 2048];
  float nr = sqrtf(x0 * x0 + x1 * x1 + x2 * x2) + 1e-6f;
  float bn = (nr - stats[c]) * stats[256 + c] * g[c] + be[c];
  float sc = bn / nr;
  float y0 = x0 * sc, y1 = x1 * sc, y2 = x2 * sc;
  out[base] = y0; out[base + 1024] = y1; out[base + 2048] = y2;
  outb[base] = f2bf(y0); outb[base + 1024] = f2bf(y1); outb[base + 2048] = f2bf(y2);
}

// ---------------------------------------------------------------------------
// VN leaky ReLU pointwise (slope = 0): out = dot>=0 ? x : x - d*dot/(|d|^2+eps)
// ---------------------------------------------------------------------------
__global__ __launch_bounds__(256) void k_vn_leaky(
    const float* __restrict__ x, const float* __restrict__ dbuf,
    unsigned short* __restrict__ outb) {
  int idx = blockIdx.x * 256 + threadIdx.x;  // over B*C*N
  int n = idx & 1023;
  int bc = idx >> 10;
  size_t base = (size_t)bc * 3072 + n;
  float x0 = x[base], x1 = x[base + 1024], x2 = x[base + 2048];
  float d0 = dbuf[base], d1 = dbuf[base + 1024], d2 = dbuf[base + 2048];
  float dot = x0 * d0 + x1 * d1 + x2 * d2;
  float dn = d0 * d0 + d1 * d1 + d2 * d2;
  float kk = dot / (dn + 1e-6f);
  bool pos = dot >= 0.f;
  float y0 = pos ? x0 : (x0 - kk * d0);
  float y1 = pos ? x1 : (x1 - kk * d1);
  float y2 = pos ? x2 : (x2 - kk * d2);
  outb[base] = f2bf(y0); outb[base + 1024] = f2bf(y1); outb[base + 2048] = f2bf(y2);
}

// ---------------------------------------------------------------------------
// Host orchestration
// ---------------------------------------------------------------------------
extern "C" void kernel_launch(void* const* d_in, const int* in_sizes, int n_in,
                              void* d_out, int out_size, void* d_ws, size_t ws_size,
                              hipStream_t stream) {
  (void)in_sizes; (void)n_in; (void)out_size; (void)ws_size;
  const float* x    = (const float*)d_in[0];
  const float* Wq   = (const float*)d_in[1];
  const float* bq   = (const float*)d_in[2];
  const float* Wk   = (const float*)d_in[3];
  const float* bk   = (const float*)d_in[4];
  const float* Wz   = (const float*)d_in[5];
  const float* bz   = (const float*)d_in[6];
  const float* Wo   = (const float*)d_in[7];
  const float* bo   = (const float*)d_in[8];
  const float* ln1g = (const float*)d_in[9];
  const float* ln1b = (const float*)d_in[10];
  const float* Wm1  = (const float*)d_in[11];
  const float* bm1  = (const float*)d_in[12];
  const float* bng  = (const float*)d_in[13];
  const float* bnb  = (const float*)d_in[14];
  const float* Wr   = (const float*)d_in[15];
  const float* Wm2  = (const float*)d_in[16];
  const float* bm2  = (const float*)d_in[17];
  const float* ln2g = (const float*)d_in[18];
  const float* ln2b = (const float*)d_in[19];
  float* out = (float*)d_out;

  // Workspace layout (with reuse): 4 bf16 slots + 3 f32 slots + small tail.
  char* wsb = (char*)d_ws;
  const size_t Hb = (size_t)8 * 256 * 3 * 1024 * 2;  // bf16 activation slot
  const size_t Fb = Hb * 2;                          // f32 activation slot
  unsigned short* H0 = (unsigned short*)(wsb);
  unsigned short* H1 = (unsigned short*)(wsb + Hb);
  unsigned short* H2 = (unsigned short*)(wsb + 2 * Hb);
  unsigned short* H3 = (unsigned short*)(wsb + 3 * Hb);
  float* F0 = (float*)(wsb + 4 * Hb);
  float* F1 = (float*)(wsb + 4 * Hb + Fb);
  float* F2 = (float*)(wsb + 4 * Hb + 2 * Fb);
  float* U  = (float*)(wsb + 4 * Hb + 3 * Fb);
  float* Uq = U + 0, *Uk = U + 768, *Uz = U + 1536;
  float* Uo = U + 2304, *Um1 = U + 3072, *Um2 = U + 3840;
  float* stats = U + 4608;

  // Bias directions eps*b/||b||.
  k_bias_dir<<<1, 256, 0, stream>>>(bq, Uq);
  k_bias_dir<<<1, 256, 0, stream>>>(bk, Uk);
  k_bias_dir<<<1, 256, 0, stream>>>(bz, Uz);
  k_bias_dir<<<1, 256, 0, stream>>>(bo, Uo);
  k_bias_dir<<<1, 256, 0, stream>>>(bm1, Um1);
  k_bias_dir<<<1, 256, 0, stream>>>(bm2, Um2);

  const int nElem = 8 * 256 * 3 * 1024;
  k_f32_to_bf16<<<nElem / 256, 256, 0, stream>>>(x, H0, nElem);  // xb

  dim3 gg(24, 2, 8);  // NC/128, M/128, batch
  // q,k,z linears (bf16 out, VN bias)
  k_gemm_vn<1, 1><<<gg, 256, 0, stream>>>(Wq, H0, H1, Uq, 256, 256, 3072);
  k_gemm_vn<1, 1><<<gg, 256, 0, stream>>>(Wk, H0, H2, Uk, 256, 256, 3072);
  k_gemm_vn<1, 1><<<gg, 256, 0, stream>>>(Wz, H0, H3, Uz, 256, 256, 3072);

  // Fused attention (softmax never hits HBM); attn out -> H0 (xb dead).
  k_flash_attn<<<dim3(16, 8, 8), 128, 0, stream>>>(H1, H2, H3, H0);

  // Output projection (f32 out).
  k_gemm_vn<0, 1><<<gg, 256, 0, stream>>>(Wo, H0, F0, Uo, 256, 256, 3072);

  // LN1 + residual x -> y2 (F1 f32 + H3 bf16).
  k_vn_ln_res<1><<<dim3(1024, 8), 256, 0, stream>>>(F0, x, ln1g, ln1b, F1, H3);

  // MLP: m1 linear (f32 out into F0).
  k_gemm_vn<0, 1><<<gg, 256, 0, stream>>>(Wm1, H3, F0, Um1, 256, 256, 3072);

  // BatchNorm (train mode) -> F2 f32 + H1 bf16.
  k_vn_bn_stats<<<256, 256, 0, stream>>>(F0, stats);
  k_vn_bn_apply<<<8192, 256, 0, stream>>>(F0, stats, bng, bnb, F2, H1);

  // Learned direction d = Wr * hbn (f32 into F0, no bias).
  k_gemm_vn<0, 0><<<gg, 256, 0, stream>>>(Wr, H1, F0, nullptr, 256, 256, 3072);

  // VN leaky ReLU (slope 0) -> H2 bf16.
  k_vn_leaky<<<8192, 256, 0, stream>>>(F2, F0, H2);

  // m2 linear (f32 into F0).
  k_gemm_vn<0, 1><<<gg, 256, 0, stream>>>(Wm2, H2, F0, Um2, 256, 256, 3072);

  // LN2 + identity(y2) -> final output.
  k_vn_ln_res<0><<<dim3(1024, 8), 256, 0, stream>>>(F0, F1, ln2g, ln2b, out, nullptr);
}
